// AgnisV5_36524401885757
// MI455X (gfx1250) — matrix operand order repaired and checked by reading, not compiled, not run on hardware
//
#include <hip/hip_runtime.h>
#include <hip/hip_bf16.h>

// ---------------- problem constants ----------------
#define Vv 50257
#define Ee 768
#define Hh 3072
#define Bb 4
#define Tt 256
#define ALPHA 0.4f
#define ETA_R 0.005f
#define LR_INFER 0.1f
#define ETA_LEARN 0.01f

typedef __attribute__((ext_vector_type(16))) __bf16 v16bf;
typedef __attribute__((ext_vector_type(8)))  float  v8f;
typedef __attribute__((ext_vector_type(4)))  unsigned int v4u;
typedef __attribute__((ext_vector_type(8)))  int    v8i;
typedef __attribute__((ext_vector_type(4)))  int    v4i;

#ifndef __has_builtin
#define __has_builtin(x) 0
#endif
#if __has_builtin(__builtin_amdgcn_tensor_load_to_lds) && \
    __has_builtin(__builtin_amdgcn_s_wait_tensorcnt)
#define USE_TDM 1
#else
#define USE_TDM 0
#endif

// round-to-nearest-even f32 -> bf16 bits
__device__ __forceinline__ unsigned short f2bf(float x) {
    unsigned u = __float_as_uint(x);
    unsigned r = (u + 0x7FFFu + ((u >> 16) & 1u)) >> 16;
    return (unsigned short)r;
}

// cluster-wide phase barrier: fence, intra-WG sync, one cluster signal per WG,
// all waves wait (barrier-done is delivered to every wave in the cluster).
__device__ __forceinline__ void cbar() {
    __threadfence();                 // release: make our stores device-visible
    __syncthreads();                 // all waves of this WG arrived
    if (threadIdx.x == 0)
        asm volatile("s_barrier_signal -3" ::: "memory");
    asm volatile("s_barrier_wait -3" ::: "memory");
    __syncthreads();
    __threadfence();                 // acquire: don't read stale lines
}

// ---------------- init: copy mutable state into workspace ----------------
__global__ __launch_bounds__(256) void k_init(
    const float* __restrict__ R0, const float* __restrict__ W1_0,
    const float* __restrict__ W2_0, const float* __restrict__ x1_0,
    const float* __restrict__ x2_0,
    float* R, float* W1, float* W2, float* h, float* x1, float* x2)
{
    size_t i0 = (size_t)blockIdx.x * blockDim.x + threadIdx.x;
    size_t st = (size_t)gridDim.x * blockDim.x;
    for (size_t i = i0; i < (size_t)Ee * Ee; i += st) R[i]  = R0[i];
    for (size_t i = i0; i < (size_t)Hh * Ee; i += st) W1[i] = W1_0[i];
    for (size_t i = i0; i < (size_t)Ee * Hh; i += st) W2[i] = W2_0[i];
    for (size_t i = i0; i < (size_t)Bb * Hh; i += st) x1[i] = x1_0[i];
    for (size_t i = i0; i < (size_t)Bb * Ee; i += st) { x2[i] = x2_0[i]; h[i] = 0.f; }
}

// ------ embedding f32 -> bf16 (read-once source streamed nontemporally; ------
// ------ the bf16 copy is the hot L2-resident GEMM operand)              ------
__global__ __launch_bounds__(256) void k_embf16(const float* __restrict__ e,
                                                unsigned short* __restrict__ o)
{
    size_t i = (size_t)blockIdx.x * blockDim.x + threadIdx.x;
    size_t st = (size_t)gridDim.x * blockDim.x;
    for (; i < (size_t)Vv * Ee; i += st)
        o[i] = f2bf(__builtin_nontemporal_load(e + i));
}

// ---------------- phase 1: persistent 16-WG cluster recurrence ----------------
__global__ __launch_bounds__(256) void k_recurrence(
    const int*   __restrict__ tok,  const float* __restrict__ embd,
    const float* __restrict__ Wc1,  const float* __restrict__ bc1,
    const float* __restrict__ Wc2,  const float* __restrict__ bc2,
    const float* __restrict__ Wg,   const float* __restrict__ bg,
    const float* __restrict__ Wt,
    const float* __restrict__ lng,  const float* __restrict__ lnb,
    float* W1, float* W2, float* R, float* h, float* x1, float* x2,
    float* f1, float* f2, float* e0, float* e1,
    float* tmpo, float* ctx, float* embn, float* core, float* cf1,
    float* cfeat, float* tfeat, float* scal,
    unsigned short* __restrict__ fusedb)
{
    const int gid = blockIdx.x * blockDim.x + threadIdx.x;
    const int NTH = gridDim.x * blockDim.x;       // 16 * 256 = 4096

    for (int t = 0; t < Tt; ++t) {
        // ---- P0: temporal = h@R ; f1,f2 = tanh(state) ; emb row inv-norms ----
        for (int m = gid; m < Bb * Ee; m += NTH) {
            int b = m / Ee, j = m % Ee;
            const float* hb = h + b * Ee;
            float s = 0.f;
            for (int k = 0; k < Ee; ++k) s += hb[k] * R[k * Ee + j];
            tmpo[m] = s;
        }
        for (int m = gid; m < Bb * Hh; m += NTH) f1[m] = tanhf(x1[m]);
        for (int m = gid; m < Bb * Ee; m += NTH) f2[m] = tanhf(x2[m]);
        if (gid < Bb) {
            int idx = tok[gid * Tt + t];
            const float* er = embd + (size_t)idx * Ee;
            float ss = 0.f;
            for (int j = 0; j < Ee; ++j) ss += er[j] * er[j];
            scal[gid] = 1.f / fmaxf(sqrtf(ss), 1e-12f);
        }
        cbar();
        // ---- P1: normalized emb + ctx ----
        for (int m = gid; m < Bb * Ee; m += NTH) {
            int b = m / Ee, j = m % Ee;
            int idx = tok[b * Tt + t];
            float ev = embd[(size_t)idx * Ee + j] * scal[b];
            embn[m] = ev;
            ctx[m]  = ev + ALPHA * tmpo[m];
        }
        cbar();
        // ---- predictive-coding inference (3 iterations) ----
        for (int it = 0; it < 3; ++it) {
            // P_a: e0 = ctx - f1@W1 ; e1 = x1 - f2@W2
            for (int m = gid; m < Bb * Ee; m += NTH) {
                int b = m / Ee, j = m % Ee;
                const float* f1b = f1 + b * Hh;
                float s = 0.f;
                for (int i = 0; i < Hh; ++i) s += f1b[i] * W1[i * Ee + j];
                e0[m] = ctx[m] - s;
            }
            for (int m = gid; m < Bb * Hh; m += NTH) {
                int b = m / Hh, i = m % Hh;
                const float* f2b = f2 + b * Ee;
                float s = 0.f;
                for (int j = 0; j < Ee; ++j) s += f2b[j] * W2[j * Hh + i];
                e1[m] = x1[m] - s;
            }
            cbar();
            // P_b: state updates + refreshed tanh (same thread owns m)
            for (int m = gid; m < Bb * Hh; m += NTH) {
                int b = m / Hh, i = m % Hh;
                const float* e0b = e0 + b * Ee;
                const float* w   = W1 + (size_t)i * Ee;
                float s = 0.f;
                for (int j = 0; j < Ee; ++j) s += e0b[j] * w[j];
                float f  = f1[m];
                float nx = x1[m] + LR_INFER * (s * (1.f - f * f) - e1[m]);
                x1[m] = nx; f1[m] = tanhf(nx);
            }
            for (int m = gid; m < Bb * Ee; m += NTH) {
                int b = m / Ee, j = m % Ee;
                const float* e1b = e1 + b * Hh;
                const float* w   = W2 + (size_t)j * Hh;
                float s = 0.f;
                for (int i = 0; i < Hh; ++i) s += e1b[i] * w[i];
                float f  = f2[m];
                float nx = x2[m] + LR_INFER * (s * (1.f - f * f));
                x2[m] = nx; f2[m] = tanhf(nx);
            }
            cbar();
        }
        // ---- P_fin: final errors + x2 row inv-norms ----
        for (int m = gid; m < Bb * Ee; m += NTH) {
            int b = m / Ee, j = m % Ee;
            const float* f1b = f1 + b * Hh;
            float s = 0.f;
            for (int i = 0; i < Hh; ++i) s += f1b[i] * W1[i * Ee + j];
            e0[m] = ctx[m] - s;
        }
        for (int m = gid; m < Bb * Hh; m += NTH) {
            int b = m / Hh, i = m % Hh;
            const float* f2b = f2 + b * Ee;
            float s = 0.f;
            for (int j = 0; j < Ee; ++j) s += f2b[j] * W2[j * Hh + i];
            e1[m] = x1[m] - s;
        }
        if (gid >= Bb && gid < 2 * Bb) {
            int b = gid - Bb;
            const float* xb = x2 + b * Ee;
            float ss = 0.f;
            for (int j = 0; j < Ee; ++j) ss += xb[j] * xb[j];
            scal[4 + b] = 1.f / fmaxf(sqrtf(ss), 1e-12f);
        }
        cbar();
        // ---- P_W: Hebbian weight updates + core = normalize(x2) ----
        for (int idx = gid; idx < Hh * Ee; idx += NTH) {
            int i = idx / Ee, j = idx % Ee;
            float s = 0.f;
            for (int b = 0; b < Bb; ++b) s += f1[b * Hh + i] * e0[b * Ee + j];
            W1[idx] += (ETA_LEARN / Bb) * s;
        }
        for (int idx = gid; idx < Ee * Hh; idx += NTH) {
            int j = idx / Hh, i = idx % Hh;
            float s = 0.f;
            for (int b = 0; b < Bb; ++b) s += f2[b * Ee + j] * e1[b * Hh + i];
            W2[idx] += (ETA_LEARN / Bb) * s;
        }
        for (int m = gid; m < Bb * Ee; m += NTH) core[m] = x2[m] * scal[4 + m / Ee];
        cbar();
        // ---- P_R: R rank-4 update + first core-feature layer ----
        for (int idx = gid; idx < Ee * Ee; idx += NTH) {
            int k = idx / Ee, j = idx % Ee;
            float s = 0.f;
            for (int b = 0; b < Bb; ++b)
                s += h[b * Ee + k] * (core[b * Ee + j] - tmpo[b * Ee + j]);
            float r = 0.999f * R[idx] + (ETA_R / Bb) * s;   // SURPRISE = 1.0
            R[idx] = fminf(fmaxf(r, -3.f), 3.f);
        }
        for (int m = gid; m < Bb * Ee; m += NTH) {
            int b = m / Ee, j = m % Ee;
            const float* cb = core + b * Ee;
            const float* w  = Wc1 + (size_t)j * Ee;
            float s = bc1[j];
            for (int k = 0; k < Ee; ++k) s += cb[k] * w[k];
            cf1[m] = 0.5f * s * (1.f + erff(s * 0.70710678118654752f)); // exact gelu
        }
        cbar();
        // ---- P_cf2: core_feat + temporal_feat ----
        for (int m = gid; m < Bb * Ee; m += NTH) {
            int b = m / Ee, j = m % Ee;
            const float* cb = cf1 + b * Ee;
            const float* w  = Wc2 + (size_t)j * Ee;
            float s = bc2[j];
            for (int k = 0; k < Ee; ++k) s += cb[k] * w[k];
            cfeat[m] = s;
        }
        for (int m = gid; m < Bb * Ee; m += NTH) {
            int b = m / Ee, j = m % Ee;
            const float* tb = tmpo + b * Ee;
            const float* w  = Wt + (size_t)j * Ee;
            float s = 0.f;
            for (int k = 0; k < Ee; ++k) s += tb[k] * w[k];
            tfeat[m] = s;
        }
        cbar();
        // ---- P_gate: gate, blend, store new h ----
        for (int m = gid; m < Bb * Ee; m += NTH) {
            int b = m / Ee, j = m % Ee;
            const float* wg = Wg + (size_t)j * (2 * Ee);
            const float* eb = embn  + b * Ee;
            const float* cb = cfeat + b * Ee;
            float z = bg[j];
            for (int k = 0; k < Ee; ++k) z += eb[k] * wg[k];
            for (int k = 0; k < Ee; ++k) z += cb[k] * wg[Ee + k];
            float g  = 1.f / (1.f + expf(-z));               // GATE_WARMUP = 1
            float ht = g * (cfeat[m] + ALPHA * tfeat[m]) + (1.f - g) * embn[m];
            h[m] = ht;
        }
        cbar();
        // ---- P_ln: per-row layernorm stats ----
        if (gid < Bb) {
            const float* hb = h + gid * Ee;
            float mu = 0.f;
            for (int j = 0; j < Ee; ++j) mu += hb[j];
            mu /= Ee;
            float v = 0.f;
            for (int j = 0; j < Ee; ++j) { float d = hb[j] - mu; v += d * d; }
            v /= Ee;
            scal[8 + gid]  = mu;
            scal[12 + gid] = 1.f / sqrtf(v + 1e-5f);
        }
        cbar();
        // ---- P_out: fused(bf16) row for the vocab GEMM ----
        for (int m = gid; m < Bb * Ee; m += NTH) {
            int b = m / Ee, j = m % Ee;
            float y = (h[m] - scal[8 + b]) * scal[12 + b] * lng[j] + lnb[j];
            fusedb[(size_t)t * Bb * Ee + m] = f2bf(y);
        }
        cbar();
    }
}

// ---------------- phase 2: logits GEMM, bf16 WMMA, f32 accumulate ----------------
// out[b,t,v] = fused[m=t*B+b,:] . emb[v,:]
// per-WG tile: 32(M) x 512(N); 8 waves = 2(M) x 4(N); per-wave 16 x 128 (8 accums)
//
// A tile staged in LDS by the Tensor Data Mover with D# padding: 16 B inserted
// every 512 B stored -> row stride 1584 B (word stride 396 = 12 mod 64 banks)
// so the 16-lane ds_load_b128 fragment reads are bank-conflict-free.
#define A_ROW_BYTES 1584

__device__ __forceinline__ int a_lds_byte(int row, int col_half) {
    int b = 2 * col_half;
    return row * A_ROW_BYTES + b + ((b >> 9) << 4);   // +16B per 512B chunk
}

__global__ __launch_bounds__(256) void k_logits(
    const unsigned short* __restrict__ fusedb,   // [1024 x 768] bf16 bits
    const unsigned short* __restrict__ embb,     // [V x 768] bf16 bits
    float* __restrict__ out)                     // [B, T, V] f32
{
    __shared__ __align__(16) unsigned short As[32 * (A_ROW_BYTES / 2)];
    const int tid = threadIdx.x;
    const int wgN = blockIdx.x;                  // 0..98
    const int wgM = blockIdx.y;                  // 0..31

#if USE_TDM
    // One wave programs the TDM: 2D tensor [1024 rows x 768 cols] of 2B elems,
    // tile = rows [wgM*32, +32) x all 768 cols, padded into LDS as above.
    if (tid < 32) {
        unsigned lds_off = (unsigned)(uintptr_t)(void*)As;
        unsigned long long gaddr =
            (unsigned long long)(uintptr_t)(fusedb + (size_t)wgM * 32 * Ee);
        v4u g0 = { 1u,                                   // count=1 (valid D#)
                   lds_off,                              // lds_addr
                   (unsigned)(gaddr & 0xFFFFFFFFull),    // global_addr[31:0]
                   (unsigned)((gaddr >> 32) & 0x1FFFFFFull) | (2u << 30) }; // +type=2
        v8i g1 = { (int)((1u << 16)    // data_size = 1 -> 2 bytes
                       | (1u << 20)    // pad_enable
                       | (6u << 22)    // pad_interval: 128 DWORDs = 512 B
                       | (3u << 25)),  // pad_amount:   4 DWORDs  = 16 B
                   (int)(768u << 16),             // tensor_dim0 = 768 (bits 79:48)
                   (int)(1024u << 16),            // tensor_dim1 = 1024 (bits 111:80)
                   (int)(768u << 16),             // tile_dim0 = 768 (bits 127:112)
                   32,                            // tile_dim1 = 32
                   768,                           // tensor_dim0_stride = 768 elems
                   0, 0 };
        v4i g2 = { 0, 0, 0, 0 };
        v4i g3 = { 0, 0, 0, 0 };
#if defined(__clang_major__) && __clang_major__ >= 23
        v8i g4 = { 0, 0, 0, 0, 0, 0, 0, 0 };
        __builtin_amdgcn_tensor_load_to_lds(g0, g1, g2, g3, g4, 0);
#else
        __builtin_amdgcn_tensor_load_to_lds(g0, g1, g2, g3, 0);
#endif
        __builtin_amdgcn_s_wait_tensorcnt(0);
    }
#else
    // Fallback: cooperative staged load into the identical padded layout.
    for (int idx = tid; idx < 32 * 96; idx += 256) {
        int row = idx / 96, c8 = (idx % 96) * 8;
        uint4 srcv = *(const uint4*)(fusedb + ((size_t)(wgM * 32 + row)) * Ee + c8);
        *(uint4*)((char*)As + a_lds_byte(row, c8)) = srcv;
    }
#endif
    __syncthreads();

    const int wave = tid >> 5, lane = tid & 31;
    const int wm = wave >> 2, wn = wave & 3;
    const int lr = lane & 15, lh = lane >> 4;
    const int arow = wm * 16 + lr;
    const int nwb  = wgN * 512 + wn * 128;
    const char* asb = (const char*)As;

    union FragU { uint4 q[2]; v16bf v; };
    v8f acc[8] = {};

    for (int kt = 0; kt < 24; ++kt) {
        const int k = kt * 32;
        // A fragment: lane holds M=arow, K = lh*8 + {0..7} and + {16..23}
        FragU a;
        a.q[0] = *(const uint4*)(asb + a_lds_byte(arow, k + lh * 8));
        a.q[1] = *(const uint4*)(asb + a_lds_byte(arow, k + lh * 8 + 16));
        // prefetch next K-slice of one emb row per lane (L2 hint)
        if (kt + 1 < 24)
            __builtin_prefetch(embb + (size_t)(nwb + lr) * Ee + k + 32, 0, 1);
#pragma unroll
        for (int s = 0; s < 8; ++s) {
            int n = nwb + s * 16 + lr;
            if (n > Vv - 1) n = Vv - 1;            // clamp tail reads in-bounds
            // B fragment: lane holds N=n, K = lh*8 + {0..7} and + {16..23}
            const unsigned short* bp = embb + (size_t)n * Ee + k + lh * 8;
            FragU bfrag;
            bfrag.q[0] = *(const uint4*)(bp);
            bfrag.q[1] = *(const uint4*)(bp + 16);
            acc[s] = __builtin_amdgcn_wmma_f32_16x16x32_bf16(
                false, a.v, false, bfrag.v, (short)0, acc[s], false, false);
        }
    }
    // D layout: lane holds col N = lr, rows r + 8*lh (r = 0..7).
    // Streaming output: nontemporal stores keep the bf16 embedding L2-resident.
#pragma unroll
    for (int s = 0; s < 8; ++s) {
        int n = nwb + s * 16 + lr;
        if (n < Vv) {
#pragma unroll
            for (int r = 0; r < 8; ++r) {
                int m = wgM * 32 + wm * 16 + r + 8 * lh;       // m = t*B + b
                __builtin_nontemporal_store(
                    acc[s][r], &out[((size_t)(m & 3) * Tt + (m >> 2)) * Vv + n]);
            }
        }
    }
}

// ---------------- host-side launch ----------------
extern "C" void kernel_launch(void* const* d_in, const int* in_sizes, int n_in,
                              void* d_out, int out_size, void* d_ws, size_t ws_size,
                              hipStream_t stream) {
    (void)in_sizes; (void)n_in; (void)out_size; (void)ws_size;
    const int*   tok  = (const int*)  d_in[0];
    const float* embd = (const float*)d_in[1];
    const float* Wc1  = (const float*)d_in[2];
    const float* bc1  = (const float*)d_in[3];
    const float* Wc2  = (const float*)d_in[4];
    const float* bc2  = (const float*)d_in[5];
    const float* Wg   = (const float*)d_in[6];
    const float* bg   = (const float*)d_in[7];
    const float* Wt   = (const float*)d_in[8];
    const float* R0   = (const float*)d_in[9];
    const float* lng  = (const float*)d_in[10];
    const float* lnb  = (const float*)d_in[11];
    const float* W1_0 = (const float*)d_in[12];
    const float* W2_0 = (const float*)d_in[13];
    const float* x1_0 = (const float*)d_in[14];
    const float* x2_0 = (const float*)d_in[15];

    char* base = (char*)d_ws;
    size_t off = 0;
    auto take = [&](size_t bytes) -> char* {
        char* p = base + off;
        off += (bytes + 255) & ~(size_t)255;
        return p;
    };
    unsigned short* embb   = (unsigned short*)take((size_t)Vv * Ee * 2); // 77.2 MB
    unsigned short* fusedb = (unsigned short*)take((size_t)Tt * Bb * Ee * 2);
    float* W1    = (float*)take((size_t)Hh * Ee * 4);
    float* W2    = (float*)take((size_t)Ee * Hh * 4);
    float* R     = (float*)take((size_t)Ee * Ee * 4);
    float* h     = (float*)take((size_t)Bb * Ee * 4);
    float* x1    = (float*)take((size_t)Bb * Hh * 4);
    float* x2    = (float*)take((size_t)Bb * Ee * 4);
    float* f1    = (float*)take((size_t)Bb * Hh * 4);
    float* f2    = (float*)take((size_t)Bb * Ee * 4);
    float* e0    = (float*)take((size_t)Bb * Ee * 4);
    float* e1    = (float*)take((size_t)Bb * Hh * 4);
    float* tmpo  = (float*)take((size_t)Bb * Ee * 4);
    float* ctx   = (float*)take((size_t)Bb * Ee * 4);
    float* embn  = (float*)take((size_t)Bb * Ee * 4);
    float* core  = (float*)take((size_t)Bb * Ee * 4);
    float* cf1   = (float*)take((size_t)Bb * Ee * 4);
    float* cfeat = (float*)take((size_t)Bb * Ee * 4);
    float* tfeat = (float*)take((size_t)Bb * Ee * 4);
    float* scal  = (float*)take(64);

    k_init<<<dim3(4096), dim3(256), 0, stream>>>(R0, W1_0, W2_0, x1_0, x2_0,
                                                 R, W1, W2, h, x1, x2);
    k_embf16<<<dim3(8192), dim3(256), 0, stream>>>(embd, embb);
    // persistent cluster kernel: 16 workgroups (one cluster's worth), 8 waves each
    k_recurrence<<<dim3(16), dim3(256), 0, stream>>>(
        tok, embd, Wc1, bc1, Wc2, bc2, Wg, bg, Wt, lng, lnb,
        W1, W2, R, h, x1, x2, f1, f2, e0, e1,
        tmpo, ctx, embn, core, cf1, cfeat, tfeat, scal, fusedb);
    dim3 g((Vv + 511) / 512, (Tt * Bb) / 32);
    k_logits<<<g, dim3(256), 0, stream>>>(fusedb, embb, (float*)d_out);
}
